// SwinTransformerBlock_13383118095102
// MI455X (gfx1250) — matrix operand order repaired and checked by reading, not compiled, not run on hardware
//
#include <hip/hip_runtime.h>
#include <hip/hip_bf16.h>
#include <math.h>

// ---------------------------------------------------------------------------
// Types & WMMA helper (CDNA5 / gfx1250, wave32)
// ---------------------------------------------------------------------------
typedef __attribute__((ext_vector_type(16))) _Float16 v16h;
typedef __attribute__((ext_vector_type(8)))  float    v8f;

union F16Frag { v16h v; uint4 q[2]; };

__device__ __forceinline__ v8f wmma_f16(v16h a, v16h b, v8f c) {
    // D = A(16x32 f16) * B(32x16 f16) + C(16x16 f32)
    return __builtin_amdgcn_wmma_f32_16x16x32_f16(false, a, false, b, (short)0, c, false, false);
}

#define M_TOK 131072      // B * H * W = 32*64*64
#define CDIM  256
#define NWIN  2048        // B * NW
#define HEADS 8
#define HD    32
#define NTOK  64          // tokens per window
#define SAPITCH 40        // padded LDS row pitch (f16) for 128b-load bank spread

// ---------------------------------------------------------------------------
// fp32 -> fp16 weight conversion
// ---------------------------------------------------------------------------
__global__ __launch_bounds__(256) void cvt_f16_kernel(const float* __restrict__ in,
                                                      _Float16* __restrict__ out, int n) {
    int i = blockIdx.x * 256 + threadIdx.x;
    if (i < n) out[i] = (_Float16)in[i];
}

// ---------------------------------------------------------------------------
// LayerNorm1 + cyclic shift + window partition  (fp32 in, f16 window-order out)
// ---------------------------------------------------------------------------
__global__ __launch_bounds__(256) void ln1_window_kernel(const float* __restrict__ x,
                                                         const float* __restrict__ w,
                                                         const float* __restrict__ b,
                                                         _Float16* __restrict__ out) {
    const int wave = threadIdx.x >> 5, lane = threadIdx.x & 31;
    const int token = blockIdx.x * 8 + wave;
    const float* xp = x + (size_t)token * CDIM;
    float vals[8]; float s = 0.f, ss = 0.f;
#pragma unroll
    for (int i = 0; i < 8; ++i) { float t = xp[lane + 32 * i]; vals[i] = t; s += t; ss += t * t; }
#pragma unroll
    for (int off = 16; off > 0; off >>= 1) { s += __shfl_xor(s, off, 32); ss += __shfl_xor(ss, off, 32); }
    const float mean = s * (1.f / 256.f);
    const float var  = ss * (1.f / 256.f) - mean * mean;
    const float rstd = rsqrtf(var + 1e-5f);
    const int bimg = token >> 12;
    const int pos  = token & 4095;
    const int h = pos >> 6, wc = pos & 63;
    const int hs  = (h + 60) & 63;       // roll(-4)
    const int ws2 = (wc + 60) & 63;
    const int dst = ((bimg * 64 + (hs >> 3) * 8 + (ws2 >> 3)) * 64) + (hs & 7) * 8 + (ws2 & 7);
    _Float16* op = out + (size_t)dst * CDIM;
#pragma unroll
    for (int i = 0; i < 8; ++i) {
        int c = lane + 32 * i;
        op[c] = (_Float16)((vals[i] - mean) * rstd * w[c] + b[c]);
    }
}

// LayerNorm2 (token order)
__global__ __launch_bounds__(256) void ln2_kernel(const float* __restrict__ x,
                                                  const float* __restrict__ w,
                                                  const float* __restrict__ b,
                                                  _Float16* __restrict__ out) {
    const int wave = threadIdx.x >> 5, lane = threadIdx.x & 31;
    const int token = blockIdx.x * 8 + wave;
    const float* xp = x + (size_t)token * CDIM;
    float vals[8]; float s = 0.f, ss = 0.f;
#pragma unroll
    for (int i = 0; i < 8; ++i) { float t = xp[lane + 32 * i]; vals[i] = t; s += t; ss += t * t; }
#pragma unroll
    for (int off = 16; off > 0; off >>= 1) { s += __shfl_xor(s, off, 32); ss += __shfl_xor(ss, off, 32); }
    const float mean = s * (1.f / 256.f);
    const float var  = ss * (1.f / 256.f) - mean * mean;
    const float rstd = rsqrtf(var + 1e-5f);
    _Float16* op = out + (size_t)token * CDIM;
#pragma unroll
    for (int i = 0; i < 8; ++i) {
        int c = lane + 32 * i;
        op[c] = (_Float16)((vals[i] - mean) * rstd * w[c] + b[c]);
    }
}

// ---------------------------------------------------------------------------
// WMMA GEMM core: block tile 128(M) x 16*NT(N), K-step 32, padded LDS rows.
// A: [M][K] f16 row-major.  Wt: [N][K] f16 row-major (torch-style weight).
// 8 waves, wave i owns rows [m0+16i, m0+16i+16), accumulating NT N-tiles.
// ---------------------------------------------------------------------------
template <int NT>
__device__ __forceinline__ void gemm_core(const _Float16* __restrict__ A,
                                          const _Float16* __restrict__ Wt,
                                          int K, int m0, int n0,
                                          _Float16* sA, _Float16* sW, v8f acc[NT]) {
    const int tid = threadIdx.x;
    const int lane = tid & 31, wave = tid >> 5;
    const int arow = tid >> 1, ahalf = tid & 1;

    for (int k0 = 0; k0 < K; k0 += 32) {
        { // A tile: 128 rows x 32 cols, 2 threads/row x 32B
            const uint4* src = (const uint4*)(A + (size_t)(m0 + arow) * K + k0 + ahalf * 16);
            uint4* dst = (uint4*)(sA + arow * SAPITCH + ahalf * 16);
            dst[0] = src[0]; dst[1] = src[1];
            if (k0 + 32 < K)   // hint next k-step tile into cache
                __builtin_prefetch(A + (size_t)(m0 + arow) * K + k0 + 32 + ahalf * 16, 0, 0);
        }
        if (arow < NT * 16) { // W tile: 16*NT rows x 32 cols
            const uint4* src = (const uint4*)(Wt + (size_t)(n0 + arow) * K + k0 + ahalf * 16);
            uint4* dst = (uint4*)(sW + arow * SAPITCH + ahalf * 16);
            dst[0] = src[0]; dst[1] = src[1];
            if (k0 + 32 < K)
                __builtin_prefetch(Wt + (size_t)(n0 + arow) * K + k0 + 32 + ahalf * 16, 0, 0);
        }
        __syncthreads();
        // A fragment: row = lane%16,
        // K = (lane/16)*16 .. +15 contiguous
        F16Frag af;
        {
            const _Float16* base = sA + (wave * 16 + (lane & 15)) * SAPITCH + ((lane >> 4) * 16);
            af.q[0] = ((const uint4*)base)[0];
            af.q[1] = ((const uint4*)base)[1];
        }
#pragma unroll
        for (int nt = 0; nt < NT; ++nt) {
            // B fragment: column = lane%16 (row of Wt), K 0..15 / 16..31 contiguous
            F16Frag bf;
            const _Float16* base = sW + (nt * 16 + (lane & 15)) * SAPITCH + ((lane >> 4) * 16);
            bf.q[0] = ((const uint4*)base)[0];
            bf.q[1] = ((const uint4*)base)[1];
            acc[nt] = wmma_f16(af.v, bf.v, acc[nt]);
        }
        __syncthreads();
    }
}

// ---- QKV: out f16 [M][768], += bias --------------------------------------
__global__ __launch_bounds__(256) void gemm_qkv_kernel(const _Float16* __restrict__ A,
                                                       const _Float16* __restrict__ Wt,
                                                       const float* __restrict__ bias,
                                                       _Float16* __restrict__ out) {
    __shared__ alignas(16) _Float16 sA[128 * SAPITCH];
    __shared__ alignas(16) _Float16 sW[64 * SAPITCH];
    const int m0 = blockIdx.x * 128, n0 = blockIdx.y * 64;
    v8f acc[4];
#pragma unroll
    for (int nt = 0; nt < 4; ++nt)
#pragma unroll
        for (int j = 0; j < 8; ++j) acc[nt][j] = 0.f;
    gemm_core<4>(A, Wt, CDIM, m0, n0, sA, sW, acc);
    const int lane = threadIdx.x & 31, wave = threadIdx.x >> 5;
    const int mbase = m0 + wave * 16 + ((lane >> 4) << 3);
#pragma unroll
    for (int nt = 0; nt < 4; ++nt) {
        int n = n0 + nt * 16 + (lane & 15);
        float bn = bias[n];
#pragma unroll
        for (int j = 0; j < 8; ++j)
            out[(size_t)(mbase + j) * 768 + n] = (_Float16)(acc[nt][j] + bn);
    }
}

// ---- proj: window-reverse + reverse-shift + residual add -> x1 fp32 ------
__global__ __launch_bounds__(256) void gemm_proj_kernel(const _Float16* __restrict__ A,
                                                        const _Float16* __restrict__ Wt,
                                                        const float* __restrict__ bias,
                                                        const float* __restrict__ xin,
                                                        float* __restrict__ x1) {
    __shared__ alignas(16) _Float16 sA[128 * SAPITCH];
    __shared__ alignas(16) _Float16 sW[64 * SAPITCH];
    const int m0 = blockIdx.x * 128, n0 = blockIdx.y * 64;
    v8f acc[4];
#pragma unroll
    for (int nt = 0; nt < 4; ++nt)
#pragma unroll
        for (int j = 0; j < 8; ++j) acc[nt][j] = 0.f;
    gemm_core<4>(A, Wt, CDIM, m0, n0, sA, sW, acc);
    const int lane = threadIdx.x & 31, wave = threadIdx.x >> 5;
    const int mbase = m0 + wave * 16 + ((lane >> 4) << 3);
#pragma unroll
    for (int nt = 0; nt < 4; ++nt) {
        int n = n0 + nt * 16 + (lane & 15);
        float bn = bias[n];
#pragma unroll
        for (int j = 0; j < 8; ++j) {
            int m = mbase + j;                       // window-order row
            int win = m >> 6, t = m & 63;
            int b = win >> 6, wloc = win & 63;
            int hs  = ((wloc >> 3) << 3) + (t >> 3); // shifted-frame coords
            int ws2 = ((wloc & 7) << 3) + (t & 7);
            int h  = (hs + 4) & 63;                  // reverse roll(+4,+4)
            int wc = (ws2 + 4) & 63;
            size_t idx = (((size_t)b << 12) + h * 64 + wc) * CDIM + n;
            x1[idx] = xin[idx] + acc[nt][j] + bn;
        }
    }
}

// ---- fc12: gate/feat paired tiles, silu(g)*f -> f16 [M][1024] -------------
__global__ __launch_bounds__(256) void gemm_fc12_kernel(const _Float16* __restrict__ A,
                                                        const _Float16* __restrict__ Wt,
                                                        const float* __restrict__ bias,
                                                        _Float16* __restrict__ out) {
    __shared__ alignas(16) _Float16 sA[128 * SAPITCH];
    __shared__ alignas(16) _Float16 sW[64 * SAPITCH];
    const int m0 = blockIdx.x * 128, n0 = blockIdx.y * 64;
    v8f accG[4], accF[4];
#pragma unroll
    for (int nt = 0; nt < 4; ++nt)
#pragma unroll
        for (int j = 0; j < 8; ++j) { accG[nt][j] = 0.f; accF[nt][j] = 0.f; }
    gemm_core<4>(A, Wt, CDIM, m0, n0, sA, sW, accG);                         // gate rows
    gemm_core<4>(A, Wt + (size_t)1024 * CDIM, CDIM, m0, n0, sA, sW, accF);   // feat rows
    const int lane = threadIdx.x & 31, wave = threadIdx.x >> 5;
    const int mbase = m0 + wave * 16 + ((lane >> 4) << 3);
#pragma unroll
    for (int nt = 0; nt < 4; ++nt) {
        int n = n0 + nt * 16 + (lane & 15);
        float bg = bias[n], bf = bias[1024 + n];
#pragma unroll
        for (int j = 0; j < 8; ++j) {
            float g = accG[nt][j] + bg;
            float f = accF[nt][j] + bf;
            float sg = g / (1.f + __expf(-g));       // silu
            out[(size_t)(mbase + j) * 1024 + n] = (_Float16)(sg * f);
        }
    }
}

// ---- fc2: K=1024, wide N tile (128) to halve HBM re-reads of hbuf ---------
__global__ __launch_bounds__(256) void gemm_fc2_kernel(const _Float16* __restrict__ A,
                                                       const _Float16* __restrict__ Wt,
                                                       const float* __restrict__ bias,
                                                       const float* __restrict__ x1,
                                                       float* __restrict__ out) {
    __shared__ alignas(16) _Float16 sA[128 * SAPITCH];
    __shared__ alignas(16) _Float16 sW[128 * SAPITCH];
    const int m0 = blockIdx.x * 128, n0 = blockIdx.y * 128;
    v8f acc[8];
#pragma unroll
    for (int nt = 0; nt < 8; ++nt)
#pragma unroll
        for (int j = 0; j < 8; ++j) acc[nt][j] = 0.f;
    gemm_core<8>(A, Wt, 1024, m0, n0, sA, sW, acc);
    const int lane = threadIdx.x & 31, wave = threadIdx.x >> 5;
    const int mbase = m0 + wave * 16 + ((lane >> 4) << 3);
#pragma unroll
    for (int nt = 0; nt < 8; ++nt) {
        int n = n0 + nt * 16 + (lane & 15);
        float bn = bias[n];
#pragma unroll
        for (int j = 0; j < 8; ++j) {
            size_t idx = (size_t)(mbase + j) * CDIM + n;
            out[idx] = x1[idx] + acc[nt][j] + bn;
        }
    }
}

// ---------------------------------------------------------------------------
// Windowed attention: one wave per (window, head).
// S = Q K^T * scale + relposbias + shiftmask ; softmax ; O = P V
// ---------------------------------------------------------------------------
#define VTPITCH 72   // padded pitch (f16) -> conflict-free b128 fragment loads

__global__ __launch_bounds__(32) void attn_kernel(const _Float16* __restrict__ qkv,
                                                  const float* __restrict__ rpb,
                                                  _Float16* __restrict__ out) {
    __shared__ alignas(16) _Float16 Vt[32 * VTPITCH];  // V transposed: [d][token]
    __shared__ alignas(16) _Float16 P [16 * VTPITCH];  // probs, current 16-row strip
    const int lane = threadIdx.x;
    const int wi = blockIdx.x >> 3;
    const int hh = blockIdx.x & 7;
    const size_t rowbase = (size_t)wi * NTOK;
    const float scale = 0.17677669529663687f;          // 32^-0.5

    // stage V transposed
#pragma unroll
    for (int ti = 0; ti < 2; ++ti) {
        int t = lane + 32 * ti;
        const _Float16* vp = qkv + (rowbase + t) * 768 + 512 + hh * HD;
#pragma unroll
        for (int d = 0; d < 32; ++d) Vt[d * VTPITCH + t] = vp[d];
    }
    // K fragments: B-layout, column = token, K = head-dim contiguous
    F16Frag kf[4];
#pragma unroll
    for (int nt = 0; nt < 4; ++nt) {
        const _Float16* kp = qkv + (rowbase + nt * 16 + (lane & 15)) * 768 + 256 + hh * HD + ((lane >> 4) * 16);
        kf[nt].q[0] = ((const uint4*)kp)[0];
        kf[nt].q[1] = ((const uint4*)kp)[1];
    }
    const int wloc = wi & 63;
    const int wy = wloc >> 3, wx = wloc & 7;
    __syncthreads();

    for (int mt = 0; mt < 4; ++mt) {
        // Q fragment (A layout)
        F16Frag qf;
        {
            const _Float16* qp = qkv + (rowbase + mt * 16 + (lane & 15)) * 768 + hh * HD + ((lane >> 4) * 16);
            qf.q[0] = ((const uint4*)qp)[0];
            qf.q[1] = ((const uint4*)qp)[1];
        }
        v8f s[4];
#pragma unroll
        for (int nt = 0; nt < 4; ++nt) {
#pragma unroll
            for (int j = 0; j < 8; ++j) s[nt][j] = 0.f;
            s[nt] = wmma_f16(qf.v, kf[nt].v, s[nt]);   // K=32 == HD, one step
        }
        // scale + relative position bias + shift mask (analytic constants)
#pragma unroll
        for (int nt = 0; nt < 4; ++nt) {
            int n = nt * 16 + (lane & 15);
            int ny = n >> 3, nx = n & 7;
            int hn = wy * 8 + ny, wn = wx * 8 + nx;
            int regn = (hn < 56 ? 0 : (hn < 60 ? 1 : 2)) * 3 + (wn < 56 ? 0 : (wn < 60 ? 1 : 2));
#pragma unroll
            for (int j = 0; j < 8; ++j) {
                int m = mt * 16 + j + ((lane >> 4) << 3);
                int my = m >> 3, mx = m & 7;
                int hm = wy * 8 + my, wm = wx * 8 + mx;
                int regm = (hm < 56 ? 0 : (hm < 60 ? 1 : 2)) * 3 + (wm < 56 ? 0 : (wm < 60 ? 1 : 2));
                int ridx = (my - ny + 7) * 15 + (mx - nx + 7);
                float bv = rpb[ridx * HEADS + hh];
                float mv = (regm == regn) ? 0.f : -100.f;
                s[nt][j] = s[nt][j] * scale + bv + mv;
            }
        }
        // row softmax: row lives in one 16-lane half across the 4 N-tiles
        float rinv[8], rmax[8];
#pragma unroll
        for (int j = 0; j < 8; ++j) {
            float t = fmaxf(fmaxf(s[0][j], s[1][j]), fmaxf(s[2][j], s[3][j]));
#pragma unroll
            for (int off = 1; off < 16; off <<= 1) t = fmaxf(t, __shfl_xor(t, off, 32));
            rmax[j] = t;
        }
#pragma unroll
        for (int nt = 0; nt < 4; ++nt)
#pragma unroll
            for (int j = 0; j < 8; ++j) s[nt][j] = __expf(s[nt][j] - rmax[j]);
#pragma unroll
        for (int j = 0; j < 8; ++j) {
            float t = s[0][j] + s[1][j] + s[2][j] + s[3][j];
#pragma unroll
            for (int off = 1; off < 16; off <<= 1) t += __shfl_xor(t, off, 32);
            rinv[j] = 1.f / t;
        }
        // write normalized P (D-layout -> LDS row-major for A re-layout)
#pragma unroll
        for (int nt = 0; nt < 4; ++nt)
#pragma unroll
            for (int j = 0; j < 8; ++j)
                P[(j + ((lane >> 4) << 3)) * VTPITCH + nt * 16 + (lane & 15)] = (_Float16)(s[nt][j] * rinv[j]);
        __syncthreads();
        // O(16x32) = P(16x64) @ V(64x32): two K=32 steps, two N-tiles
        v8f o0, o1;
#pragma unroll
        for (int j = 0; j < 8; ++j) { o0[j] = 0.f; o1[j] = 0.f; }
#pragma unroll
        for (int kk = 0; kk < 2; ++kk) {
            F16Frag pf;
            const _Float16* pp = P + (lane & 15) * VTPITCH + kk * 32 + ((lane >> 4) * 16);
            pf.q[0] = ((const uint4*)pp)[0];
            pf.q[1] = ((const uint4*)pp)[1];
            F16Frag vf0, vf1;
            const _Float16* vp0 = Vt + (lane & 15) * VTPITCH + kk * 32 + ((lane >> 4) * 16);
            const _Float16* vp1 = Vt + (16 + (lane & 15)) * VTPITCH + kk * 32 + ((lane >> 4) * 16);
            vf0.q[0] = ((const uint4*)vp0)[0]; vf0.q[1] = ((const uint4*)vp0)[1];
            vf1.q[0] = ((const uint4*)vp1)[0]; vf1.q[1] = ((const uint4*)vp1)[1];
            o0 = wmma_f16(pf.v, vf0.v, o0);
            o1 = wmma_f16(pf.v, vf1.v, o1);
        }
        // store head output: channel = head*32 + d, window-token row order
#pragma unroll
        for (int j = 0; j < 8; ++j) {
            int m = mt * 16 + j + ((lane >> 4) << 3);
            _Float16* op = out + (rowbase + m) * CDIM + hh * HD;
            op[lane & 15] = (_Float16)o0[j];
            op[16 + (lane & 15)] = (_Float16)o1[j];
        }
        __syncthreads();
    }
}

// ---------------------------------------------------------------------------
// launch
// ---------------------------------------------------------------------------
extern "C" void kernel_launch(void* const* d_in, const int* in_sizes, int n_in,
                              void* d_out, int out_size, void* d_ws, size_t ws_size,
                              hipStream_t stream) {
    (void)in_sizes; (void)n_in; (void)out_size; (void)ws_size;
    const float* x      = (const float*)d_in[0];
    const float* n1w    = (const float*)d_in[1];
    const float* n1b    = (const float*)d_in[2];
    const float* qkv_w  = (const float*)d_in[3];
    const float* qkv_b  = (const float*)d_in[4];
    const float* proj_w = (const float*)d_in[5];
    const float* proj_b = (const float*)d_in[6];
    const float* rpb    = (const float*)d_in[7];
    const float* n2w    = (const float*)d_in[8];
    const float* n2b    = (const float*)d_in[9];
    const float* fc12_w = (const float*)d_in[10];
    const float* fc12_b = (const float*)d_in[11];
    const float* fc2_w  = (const float*)d_in[12];
    const float* fc2_b  = (const float*)d_in[13];
    float* out = (float*)d_out;

    char* ws = (char*)d_ws;
    size_t off = 0;
    auto take = [&](size_t bytes) -> char* {
        char* p = ws + off;
        off += (bytes + 255) & ~(size_t)255;
        return p;
    };
    const size_t M = M_TOK;
    // regionA: xwin (f16) -> attn_out (f16) -> ln2 (f16), aliased across phases
    _Float16* regA = (_Float16*)take(M * CDIM * 2);
    // regionB: qkv (f16 [M][768]) -> hidden (f16 [M][1024]), aliased
    _Float16* regB = (_Float16*)take(M * 2048 * 2);
    float*    x1   = (float*)take(M * CDIM * 4);
    _Float16* wq   = (_Float16*)take((size_t)768 * 256 * 2);
    _Float16* wp   = (_Float16*)take((size_t)256 * 256 * 2);
    _Float16* w12  = (_Float16*)take((size_t)2048 * 256 * 2);
    _Float16* w2   = (_Float16*)take((size_t)256 * 1024 * 2);

    cvt_f16_kernel<<<(768 * 256 + 255) / 256, 256, 0, stream>>>(qkv_w, wq, 768 * 256);
    cvt_f16_kernel<<<(256 * 256 + 255) / 256, 256, 0, stream>>>(proj_w, wp, 256 * 256);
    cvt_f16_kernel<<<(2048 * 256 + 255) / 256, 256, 0, stream>>>(fc12_w, w12, 2048 * 256);
    cvt_f16_kernel<<<(256 * 1024 + 255) / 256, 256, 0, stream>>>(fc2_w, w2, 256 * 1024);

    ln1_window_kernel<<<M / 8, 256, 0, stream>>>(x, n1w, n1b, regA);
    gemm_qkv_kernel<<<dim3(M / 128, 768 / 64), 256, 0, stream>>>(regA, wq, qkv_b, regB);
    attn_kernel<<<NWIN * HEADS, 32, 0, stream>>>(regB, rpb, regA);
    gemm_proj_kernel<<<dim3(M / 128, 256 / 64), 256, 0, stream>>>(regA, wp, proj_b, x, x1);
    ln2_kernel<<<M / 8, 256, 0, stream>>>(x1, n2w, n2b, regA);
    gemm_fc12_kernel<<<dim3(M / 128, 1024 / 64), 256, 0, stream>>>(regA, w12, fc12_b, regB);
    gemm_fc2_kernel<<<dim3(M / 128, 256 / 128), 256, 0, stream>>>(regB, w2, fc2_b, x1, out);
}
